// QuantumNetwork_69252052681206
// MI455X (gfx1250) — matrix operand order repaired and checked by reading, not compiled, not run on hardware
//
#include <hip/hip_runtime.h>
#include <cmath>

// ---------------------------------------------------------------------------
// Types for CDNA5 WMMA (gfx1250, wave32)
// ---------------------------------------------------------------------------
typedef __bf16 bf16_t;
typedef __attribute__((ext_vector_type(16))) __bf16 v16bf;
typedef __attribute__((ext_vector_type(8)))  __bf16 v8bf;
typedef __attribute__((ext_vector_type(8)))  float  v8f;

// global (addrspace 1) pointer types: force global_load_b128 (LOADcnt only),
// not flat_load_b128 (which also ties up DScnt per ISA 10.1.1)
typedef __attribute__((address_space(1))) const bf16_t gc_bf16;
typedef __attribute__((address_space(1))) const v16bf  gc_v16bf;

#define WMMA_BF16(A_, B_, C_) \
  __builtin_amdgcn_wmma_f32_16x16x32_bf16(false, (A_), false, (B_), (short)0, (C_), false, false)

// A-fragment (16x32 bf16, row = lane&15): per lane two 16B chunks:
//   K = khalf*16 .. +7  (VGPRs 0-3)   and   K = khalf*16+8 .. +7 (VGPRs 4-7)
__device__ __forceinline__ v16bf load_a_frag(const bf16_t* p) {
  v8bf lo = *(const v8bf*)(p);
  v8bf hi = *(const v8bf*)(p + 8);
  return __builtin_shufflevector(lo, hi, 0,1,2,3,4,5,6,7,8,9,10,11,12,13,14,15);
}

// ---------------------------------------------------------------------------
// Problem constants (from reference)
// ---------------------------------------------------------------------------
constexpr int BATCH = 512;
constexpr int TT    = 256;   // time steps
constexpr int DIN   = 128;
constexpr int HH    = 256;   // hidden
constexpr int KC    = 384;   // DIN + HH (combined LSTM K)
constexpr int EA    = 320;   // embed E = H + F
constexpr int FF    = 64;    // s features
constexpr int DMRG  = 384;   // merged_state = E + F
constexpr int NQ    = 10;    // qubits
constexpr int NAMP  = 1024;  // 2^NQ
constexpr int NL1   = 512;
constexpr int K1P   = 416;   // (DMRG + NQ = 394) padded to mult of 32
constexpr float PI_F = 3.14159265358979323846f;

// ---------------------------------------------------------------------------
// Weight prep: fp32 -> bf16 (+ concat / pad)
// ---------------------------------------------------------------------------
__global__ void k_convert(const float* __restrict__ src, bf16_t* __restrict__ dst, int n) {
  int i = blockIdx.x * blockDim.x + threadIdx.x;
  if (i < n) dst[i] = (bf16_t)src[i];
}

__global__ void k_wcomb(const float* __restrict__ wih, const float* __restrict__ whh,
                        bf16_t* __restrict__ dst) {
  int i = blockIdx.x * blockDim.x + threadIdx.x;
  if (i < 1024 * KC) {
    int n = i / KC, k = i % KC;
    float v = (k < DIN) ? wih[n * DIN + k] : whh[n * HH + (k - DIN)];
    dst[i] = (bf16_t)v;
  }
}

__global__ void k_w1pad(const float* __restrict__ w1, bf16_t* __restrict__ dst) {
  int i = blockIdx.x * blockDim.x + threadIdx.x;
  if (i < NL1 * K1P) {
    int n = i / K1P, k = i % K1P;
    dst[i] = (k < DMRG + NQ) ? (bf16_t)w1[n * (DMRG + NQ) + k] : (bf16_t)0.f;
  }
}

// ---------------------------------------------------------------------------
// LSTM: 32 blocks x 256 thr (8 waves). Block = 16 batch rows, all 1024 gate
// cols. Wave w owns hidden cols [32w, 32w+32): all 4 gates for those units
// stay in this wave's accumulators -> cell update needs no LDS round trip.
// A-panel [x_t | h] lives in LDS as bf16; weights stream from L2 every step
// (opaque base pointer stops the compiler hoisting 96 B-frags into scratch;
// addrspace(1) cast keeps the loads on the global path / LOADcnt only).
// ---------------------------------------------------------------------------
__global__ __launch_bounds__(256) void k_lstm(
    const float* __restrict__ x, const float* __restrict__ bih, const float* __restrict__ bhh,
    const bf16_t* __restrict__ wcomb, bf16_t* __restrict__ hs, float* __restrict__ lastf)
{
  __shared__ alignas(32) bf16_t sA[16 * KC];   // 12 KB
  const int tid = threadIdx.x, lane = tid & 31, w = tid >> 5;
  const int b0 = blockIdx.x * 16;
  const int mloc = lane & 15, khalf = lane >> 4, mb = khalf * 8;

  // h := 0
  for (int i = tid; i < 16 * HH; i += 256) {
    int m = i >> 8, j = i & 255;
    sA[m * KC + DIN + j] = (bf16_t)0.f;
  }

  // per-lane weight-row element offsets and biases: n = g*256 + (2w+jt)*16 + mloc
  size_t rowoff[2][4];
  float bs[2][4];
  #pragma unroll
  for (int jt = 0; jt < 2; ++jt)
    #pragma unroll
    for (int g = 0; g < 4; ++g) {
      int n = g * 256 + (2 * w + jt) * 16 + mloc;
      rowoff[jt][g] = (size_t)n * KC;
      bs[jt][g] = bih[n] + bhh[n];
    }

  float cst[2][8];
  #pragma unroll
  for (int jt = 0; jt < 2; ++jt)
    #pragma unroll
    for (int r = 0; r < 8; ++r) cst[jt][r] = 0.f;

  const v8f zerov = {};
  uintptr_t wbase = (uintptr_t)wcomb;

  for (int t = 0; t < TT; ++t) {
    // opaque redefinition: compiler cannot hoist weight loads out of the t-loop
    asm volatile("" : "+s"(wbase));
    gc_bf16* wct = (gc_bf16*)wbase;

    // stage x_t (coalesced) as bf16 into A-panel cols [0,128)
    for (int i = tid; i < 16 * DIN; i += 256) {
      int m = i >> 7, k = i & 127;
      sA[m * KC + k] = (bf16_t)x[((size_t)(b0 + m) * TT + t) * DIN + k];
    }
    // prefetch next step's x rows (4 x 128B lines per 512B row)
    if (t + 1 < TT && tid < 64) {
      int m = tid >> 2, ln = tid & 3;
      __builtin_prefetch(&x[((size_t)(b0 + m) * TT + (t + 1)) * DIN + ln * 32], 0, 0);
    }
    __syncthreads();

    v8f acc[2][4];
    #pragma unroll
    for (int jt = 0; jt < 2; ++jt)
      #pragma unroll
      for (int g = 0; g < 4; ++g) acc[jt][g] = zerov;

    for (int kk = 0; kk < KC / 32; ++kk) {
      v16bf a = load_a_frag(&sA[mloc * KC + kk * 32 + khalf * 16]);
      int kb = kk * 32 + khalf * 16;
      v16bf bfr[2][4];
      #pragma unroll
      for (int jt = 0; jt < 2; ++jt)
        #pragma unroll
        for (int g = 0; g < 4; ++g)
          bfr[jt][g] = *(gc_v16bf*)(wct + rowoff[jt][g] + kb);
      #pragma unroll
      for (int jt = 0; jt < 2; ++jt)
        #pragma unroll
        for (int g = 0; g < 4; ++g)
          acc[jt][g] = WMMA_BF16(a, bfr[jt][g], acc[jt][g]);
    }
    __syncthreads();   // all waves finished reading h-panel before we overwrite it

    #pragma unroll
    for (int jt = 0; jt < 2; ++jt) {
      int j = (2 * w + jt) * 16 + mloc;
      #pragma unroll
      for (int r = 0; r < 8; ++r) {
        float gi = acc[jt][0][r] + bs[jt][0];
        float gf = acc[jt][1][r] + bs[jt][1];
        float gg = acc[jt][2][r] + bs[jt][2];
        float go = acc[jt][3][r] + bs[jt][3];
        float iv = 1.f / (1.f + __expf(-gi));
        float fv = 1.f / (1.f + __expf(-gf));
        float gv = tanhf(gg);
        float ov = 1.f / (1.f + __expf(-go));
        float c  = fv * cst[jt][r] + iv * gv;
        cst[jt][r] = c;
        float h = ov * tanhf(c);
        int m = mb + r;
        sA[m * KC + DIN + j] = (bf16_t)h;                       // feed next step
        hs[((size_t)(b0 + m) * TT + t) * HH + j] = (bf16_t)h;   // for proj/attn
        if (t == TT - 1) lastf[(size_t)(b0 + m) * HH + j] = h;
      }
    }
    __syncthreads();
  }
}

// ---------------------------------------------------------------------------
// Fused proj -> k/v -> attention (online softmax over T), then out_proj,
// merged_state and state_norm. 32 blocks x 256 thr, 16 batch rows per block.
// k/v never touch HBM. ctx kept in registers (20 floats/thread).
// ---------------------------------------------------------------------------
__global__ __launch_bounds__(256) void k_attn(
    const float* __restrict__ svec,       // s (B,64)
    const float* __restrict__ lastf,      // (B,256)
    const bf16_t* __restrict__ projW,     // (320,256) bf16
    const float* __restrict__ proj_b,
    const bf16_t* __restrict__ inpW,      // (960,320) bf16: Wq|Wk|Wv
    const float* __restrict__ inp_b,
    const bf16_t* __restrict__ outW,      // (320,320) bf16
    const float* __restrict__ out_b,
    const bf16_t* __restrict__ hs,        // (B,T,256) bf16
    float* __restrict__ merged,           // (B,384)
    float* __restrict__ snorm)            // (B,384)
{
  __shared__ alignas(32) bf16_t sIn[16 * EA];  // state_full / hs_t / final ctx
  __shared__ alignas(32) bf16_t sP[16 * EA];
  __shared__ alignas(32) bf16_t sK[16 * EA];
  __shared__ alignas(32) bf16_t sV[16 * EA];
  __shared__ alignas(32) bf16_t sQ[16 * EA];
  __shared__ float sPp[16], sAl[16], sMr[16], sLr[16];

  const int tid = threadIdx.x, lane = tid & 31, w = tid >> 5;
  const int b0 = blockIdx.x * 16;
  const int mloc = lane & 15, khalf = lane >> 4, mb = khalf * 8;
  const float scale = 0.05590169943749474f;   // 1/sqrt(320)
  const v8f zerov = {};

  // state_full = [lstm_last | s]
  for (int i = tid; i < 16 * EA; i += 256) {
    int m = i / EA, e = i % EA;
    float v = (e < HH) ? lastf[(size_t)(b0 + m) * HH + e]
                       : svec[(size_t)(b0 + m) * FF + (e - HH)];
    sIn[i] = (bf16_t)v;
  }
  __syncthreads();

  // q = state_full @ Wq^T + bq   (rows 0..319 of in_proj)
  for (int nt = w; nt < EA / 16; nt += 8) {
    int n = nt * 16 + mloc;
    v8f acc = zerov;
    const bf16_t* arow = sIn + mloc * EA;
    const bf16_t* brow = inpW + (size_t)n * EA;
    for (int kk = 0; kk < EA / 32; ++kk) {
      v16bf a = load_a_frag(arow + kk * 32 + khalf * 16);
      v16bf b = *(const v16bf*)(brow + kk * 32 + khalf * 16);
      acc = WMMA_BF16(a, b, acc);
    }
    float bias = inp_b[n];
    #pragma unroll
    for (int r = 0; r < 8; ++r) sQ[(mb + r) * EA + n] = (bf16_t)(acc[r] + bias);
  }
  if (tid < 16) { sMr[tid] = -1e30f; sLr[tid] = 0.f; }

  float ctx[20];
  #pragma unroll
  for (int r = 0; r < 20; ++r) ctx[r] = 0.f;
  const int cm = tid & 15;        // ctx row; cols e = (tid>>4) + 16*r
  const int ce = tid >> 4;

  uintptr_t pwb = (uintptr_t)projW;
  uintptr_t iwb = (uintptr_t)inpW;
  __syncthreads();

  for (int t = 0; t < TT; ++t) {
    // opaque: keep weight loads inside the loop (L2-resident), not scratch-spilled
    asm volatile("" : "+s"(pwb), "+s"(iwb));
    gc_bf16* pw = (gc_bf16*)pwb;
    gc_bf16* iw = (gc_bf16*)iwb;

    // hs_t (already bf16) -> sIn[:, 0:256)
    for (int i = tid; i < 16 * HH; i += 256) {
      int m = i >> 8, j = i & 255;
      sIn[m * EA + j] = hs[((size_t)(b0 + m) * TT + t) * HH + j];
    }
    if (t + 1 < TT && tid < 64) {          // prefetch next hs tile (4x128B per row)
      int m = tid >> 2, ln = tid & 3;
      __builtin_prefetch(&hs[((size_t)(b0 + m) * TT + (t + 1)) * HH + ln * 64], 0, 0);
    }
    __syncthreads();

    // proj_t = hs_t @ projW^T + pb  (K=256)
    for (int nt = w; nt < EA / 16; nt += 8) {
      int n = nt * 16 + mloc;
      v8f acc = zerov;
      const bf16_t* arow = sIn + mloc * EA;
      gc_bf16* brow = pw + (size_t)n * HH;
      for (int kk = 0; kk < HH / 32; ++kk) {
        v16bf a = load_a_frag(arow + kk * 32 + khalf * 16);
        v16bf b = *(gc_v16bf*)(brow + kk * 32 + khalf * 16);
        acc = WMMA_BF16(a, b, acc);
      }
      float bias = proj_b[n];
      #pragma unroll
      for (int r = 0; r < 8; ++r) sP[(mb + r) * EA + n] = (bf16_t)(acc[r] + bias);
    }
    __syncthreads();

    // k_t and v_t  (40 tiles across 8 waves)
    for (int tt2 = w; tt2 < 40; tt2 += 8) {
      int isv = (tt2 >= 20) ? 1 : 0;
      int nt = isv ? tt2 - 20 : tt2;
      int n = nt * 16 + mloc;
      int browi = (isv ? 2 * EA : EA) + n;     // Wk rows 320.., Wv rows 640..
      v8f acc = zerov;
      const bf16_t* arow = sP + mloc * EA;
      gc_bf16* brow = iw + (size_t)browi * EA;
      for (int kk = 0; kk < EA / 32; ++kk) {
        v16bf a = load_a_frag(arow + kk * 32 + khalf * 16);
        v16bf b = *(gc_v16bf*)(brow + kk * 32 + khalf * 16);
        acc = WMMA_BF16(a, b, acc);
      }
      float bias = inp_b[browi];
      bf16_t* dst = isv ? sV : sK;
      #pragma unroll
      for (int r = 0; r < 8; ++r) dst[(mb + r) * EA + n] = (bf16_t)(acc[r] + bias);
    }
    __syncthreads();

    // scores: 16 threads per row, butterfly-reduce within 16-lane groups
    {
      const int row = tid >> 4, sub = tid & 15;
      float d = 0.f;
      #pragma unroll
      for (int r = 0; r < 20; ++r) {
        int e = sub + 16 * r;
        d += (float)sQ[row * EA + e] * (float)sK[row * EA + e];
      }
      d += __shfl_xor(d, 1, 32);
      d += __shfl_xor(d, 2, 32);
      d += __shfl_xor(d, 4, 32);
      d += __shfl_xor(d, 8, 32);
      if (sub == 0) {                       // one writer per row: online softmax
        float sc = d * scale;
        float mo = sMr[row];
        float mn = fmaxf(mo, sc);
        float al = __expf(mo - mn);
        float p  = __expf(sc - mn);
        sLr[row] = sLr[row] * al + p;
        sMr[row] = mn;
        sAl[row] = al;
        sPp[row] = p;
      }
    }
    __syncthreads();

    // ctx = ctx*alpha + p * v_t
    {
      float al = sAl[cm], p = sPp[cm];
      #pragma unroll
      for (int r = 0; r < 20; ++r) {
        int e = ce + 16 * r;
        ctx[r] = ctx[r] * al + p * (float)sV[cm * EA + e];
      }
    }
    __syncthreads();
  }

  // normalize ctx, stage as bf16 A-panel (reuse sIn)
  {
    float inv = 1.f / sLr[cm];
    #pragma unroll
    for (int r = 0; r < 20; ++r) {
      int e = ce + 16 * r;
      sIn[cm * EA + e] = (bf16_t)(ctx[r] * inv);
    }
  }
  __syncthreads();

  // attn_out = ctx @ outW^T + ob -> merged_state cols [0,320) + state_norm
  for (int nt = w; nt < EA / 16; nt += 8) {
    int n = nt * 16 + mloc;
    v8f acc = zerov;
    const bf16_t* arow = sIn + mloc * EA;
    const bf16_t* brow = outW + (size_t)n * EA;
    for (int kk = 0; kk < EA / 32; ++kk) {
      v16bf a = load_a_frag(arow + kk * 32 + khalf * 16);
      v16bf b = *(const v16bf*)(brow + kk * 32 + khalf * 16);
      acc = WMMA_BF16(a, b, acc);
    }
    float bias = out_b[n];
    #pragma unroll
    for (int r = 0; r < 8; ++r) {
      float v = acc[r] + bias;
      size_t idx = (size_t)(b0 + mb + r) * DMRG + n;
      merged[idx] = v;
      snorm[idx]  = tanhf(v) * PI_F;
    }
  }
  // merged cols [320,384) = s
  for (int i = tid; i < 16 * FF; i += 256) {
    int m = i >> 6, e = i & 63;
    float v = svec[(size_t)(b0 + m) * FF + e];
    size_t idx = (size_t)(b0 + m) * DMRG + EA + e;
    merged[idx] = v;
    snorm[idx]  = tanhf(v) * PI_F;
  }
}

// ---------------------------------------------------------------------------
// VQC statevector sim: 512 blocks (one per batch row) x 256 thr.
// 1024 complex amplitudes in LDS; qubit q <-> bit (9-q) of flat index.
// ---------------------------------------------------------------------------
__device__ __forceinline__ void gate1q(float* re, float* im, int q, int kind,
                                       float theta, int tid) {
  float c = __cosf(0.5f * theta), s = __sinf(0.5f * theta);
  int bit = 1 << (9 - q);
  int lowm = bit - 1;
  for (int k = tid; k < NAMP / 2; k += 256) {
    int i0 = ((k & ~lowm) << 1) | (k & lowm);
    int i1 = i0 | bit;
    float a0r = re[i0], a0i = im[i0], a1r = re[i1], a1i = im[i1];
    float n0r, n0i, n1r, n1i;
    if (kind == 0) {        // RX: [[c,-is],[-is,c]]
      n0r = c * a0r + s * a1i;  n0i = c * a0i - s * a1r;
      n1r = c * a1r + s * a0i;  n1i = c * a1i - s * a0r;
    } else if (kind == 1) { // RY: [[c,-s],[s,c]]
      n0r = c * a0r - s * a1r;  n0i = c * a0i - s * a1i;
      n1r = s * a0r + c * a1r;  n1i = s * a0i + c * a1i;
    } else {                // RZ: diag(e^{-it/2}, e^{it/2})
      n0r = c * a0r + s * a0i;  n0i = c * a0i - s * a0r;
      n1r = c * a1r - s * a1i;  n1i = c * a1i + s * a1r;
    }
    re[i0] = n0r; im[i0] = n0i; re[i1] = n1r; im[i1] = n1i;
  }
  __syncthreads();
}

__device__ __forceinline__ void cnotg(float* re, float* im, int ctrl, int tgt, int tid) {
  int pc = 9 - ctrl, pt = 9 - tgt;
  int p1 = (pc < pt) ? pc : pt;
  int p2 = (pc < pt) ? pt : pc;
  int m1 = (1 << p1) - 1;
  int m2 = (1 << (p2 - 1)) - 1;
  int k = tid;                       // 256 pairs, 256 threads
  int low  = k & m1;
  int mid  = (k & ~m1 & m2) << 1;
  int high = (k & ~m2) << 2;
  int i0 = (high | mid | low) | (1 << pc);   // ctrl=1, tgt=0
  int i1 = i0 | (1 << pt);
  float tr = re[i0]; re[i0] = re[i1]; re[i1] = tr;
  float ti = im[i0]; im[i0] = im[i1]; im[i1] = ti;
  __syncthreads();
}

__global__ __launch_bounds__(256) void k_vqc(const float* __restrict__ snorm,
                                             const float* __restrict__ qw,
                                             float* __restrict__ qfeat) {
  __shared__ float re[NAMP], im[NAMP];
  __shared__ float ang[40];
  __shared__ float ez[16];
  const int b = blockIdx.x, tid = threadIdx.x;

  for (int i = tid; i < NAMP; i += 256) { re[i] = 0.03125f; im[i] = 0.f; }
  if (tid < 40) ang[tid] = snorm[(size_t)b * DMRG + tid];  // (layer*10+q)%384 == layer*10+q
  if (tid < 16) ez[tid] = 0.f;
  __syncthreads();

  for (int layer = 0; layer < 4; ++layer) {
    for (int q = 0; q < NQ; ++q) {
      float a = ang[layer * NQ + q];
      const float* wq = qw + (layer * NQ + q) * 3;
      gate1q(re, im, q, 0, a, tid);       // RX(ang)
      gate1q(re, im, q, 1, a, tid);       // RY(ang)
      gate1q(re, im, q, 2, wq[0], tid);   // RZ(w0)
      gate1q(re, im, q, 1, wq[1], tid);   // RY(w1)
      gate1q(re, im, q, 2, wq[2], tid);   // RZ(w2)
    }
    for (int q = 0; q < NQ - 1; ++q) cnotg(re, im, q, q + 1, tid);
    cnotg(re, im, NQ - 1, 0, tid);
  }

  float part[NQ];
  #pragma unroll
  for (int q = 0; q < NQ; ++q) part[q] = 0.f;
  for (int i = tid; i < NAMP; i += 256) {
    float p = re[i] * re[i] + im[i] * im[i];
    #pragma unroll
    for (int q = 0; q < NQ; ++q)
      part[q] += (i & (1 << (9 - q))) ? -p : p;
  }
  #pragma unroll
  for (int q = 0; q < NQ; ++q) atomicAdd(&ez[q], part[q]);   // ds_add_f32
  __syncthreads();
  if (tid < NQ) qfeat[(size_t)b * 16 + tid] = ez[tid];
}

// ---------------------------------------------------------------------------
// MLP head
// ---------------------------------------------------------------------------
__global__ void k_pack(const float* __restrict__ merged, const float* __restrict__ qfeat,
                       bf16_t* __restrict__ dst) {
  int i = blockIdx.x * blockDim.x + threadIdx.x;
  if (i < BATCH * K1P) {
    int n = i / K1P, k = i % K1P;
    float v = (k < DMRG) ? merged[(size_t)n * DMRG + k]
            : (k < DMRG + NQ) ? qfeat[(size_t)n * 16 + (k - DMRG)] : 0.f;
    dst[i] = (bf16_t)v;
  }
}

__global__ __launch_bounds__(256) void k_mlp1(const bf16_t* __restrict__ A,
                                              const bf16_t* __restrict__ W1p,
                                              const float* __restrict__ b1,
                                              bf16_t* __restrict__ hid) {
  const int lane = threadIdx.x & 31, w = threadIdx.x >> 5;
  const int tile = blockIdx.x * 8 + w;           // 1024 tiles (32x32 of 16x16)
  const int mt = tile >> 5, nt = tile & 31;
  const int mloc = lane & 15, khalf = lane >> 4, mb = khalf * 8;
  v8f acc = {};
  const bf16_t* arow = A   + (size_t)(mt * 16 + mloc) * K1P;
  const bf16_t* brow = W1p + (size_t)(nt * 16 + mloc) * K1P;
  for (int kk = 0; kk < K1P / 32; ++kk) {
    v16bf a = load_a_frag(arow + kk * 32 + khalf * 16);
    v16bf b = *(const v16bf*)(brow + kk * 32 + khalf * 16);
    acc = WMMA_BF16(a, b, acc);
  }
  int n = nt * 16 + mloc;
  float bias = b1[n];
  #pragma unroll
  for (int r = 0; r < 8; ++r) {
    float v = acc[r] + bias;
    hid[(size_t)(mt * 16 + mb + r) * NL1 + n] = (bf16_t)fmaxf(v, 0.f);
  }
}

__global__ void k_mlp2(const bf16_t* __restrict__ hid, const float* __restrict__ W2,
                       const float* __restrict__ b2, float* __restrict__ out) {
  int g = blockIdx.x * blockDim.x + threadIdx.x;
  if (g < BATCH * 8) {
    int bb = g >> 3, o = g & 7;
    float acc = b2[o];
    const bf16_t* h = hid + (size_t)bb * NL1;
    const float* wr = W2 + (size_t)o * NL1;
    for (int k = 0; k < NL1; ++k) acc += (float)h[k] * wr[k];
    out[g] = acc;
  }
}

// ---------------------------------------------------------------------------
// Launch
// ---------------------------------------------------------------------------
extern "C" void kernel_launch(void* const* d_in, const int* in_sizes, int n_in,
                              void* d_out, int out_size, void* d_ws, size_t ws_size,
                              hipStream_t stream) {
  (void)in_sizes; (void)n_in; (void)out_size; (void)ws_size;
  const float* s      = (const float*)d_in[0];
  const float* x      = (const float*)d_in[1];
  const float* W_ih   = (const float*)d_in[2];
  const float* W_hh   = (const float*)d_in[3];
  const float* b_ih   = (const float*)d_in[4];
  const float* b_hh   = (const float*)d_in[5];
  const float* proj_W = (const float*)d_in[6];
  const float* proj_b = (const float*)d_in[7];
  const float* inp_W  = (const float*)d_in[8];
  const float* inp_b  = (const float*)d_in[9];
  const float* out_W  = (const float*)d_in[10];
  const float* out_b  = (const float*)d_in[11];
  const float* qw     = (const float*)d_in[12];
  const float* W1     = (const float*)d_in[13];
  const float* b1     = (const float*)d_in[14];
  const float* W2     = (const float*)d_in[15];
  const float* b2     = (const float*)d_in[16];
  float* out = (float*)d_out;

  // workspace layout (bytes), all 256B-aligned
  char* ws = (char*)d_ws;
  size_t off = 0;
  auto take = [&](size_t bytes) { size_t o = off; off += (bytes + 255) & ~(size_t)255; return o; };
  bf16_t* wcomb  = (bf16_t*)(ws + take((size_t)1024 * KC * 2));
  bf16_t* projWb = (bf16_t*)(ws + take((size_t)EA * HH * 2));
  bf16_t* inpWb  = (bf16_t*)(ws + take((size_t)3 * EA * EA * 2));
  bf16_t* outWb  = (bf16_t*)(ws + take((size_t)EA * EA * 2));
  bf16_t* w1p    = (bf16_t*)(ws + take((size_t)NL1 * K1P * 2));
  bf16_t* hs     = (bf16_t*)(ws + take((size_t)BATCH * TT * HH * 2));
  float*  lastf  = (float*) (ws + take((size_t)BATCH * HH * 4));
  float*  merged = (float*) (ws + take((size_t)BATCH * DMRG * 4));
  float*  snorm  = (float*) (ws + take((size_t)BATCH * DMRG * 4));
  float*  qfeat  = (float*) (ws + take((size_t)BATCH * 16 * 4));
  bf16_t* mpack  = (bf16_t*)(ws + take((size_t)BATCH * K1P * 2));
  bf16_t* hid    = (bf16_t*)(ws + take((size_t)BATCH * NL1 * 2));

  // 1. weight prep (fp32 -> bf16, concat, pad)
  k_convert<<<(EA * HH + 255) / 256, 256, 0, stream>>>(proj_W, projWb, EA * HH);
  k_convert<<<(3 * EA * EA + 255) / 256, 256, 0, stream>>>(inp_W, inpWb, 3 * EA * EA);
  k_convert<<<(EA * EA + 255) / 256, 256, 0, stream>>>(out_W, outWb, EA * EA);
  k_wcomb<<<(1024 * KC + 255) / 256, 256, 0, stream>>>(W_ih, W_hh, wcomb);
  k_w1pad<<<(NL1 * K1P + 255) / 256, 256, 0, stream>>>(W1, w1p);

  // 2. LSTM over T (WMMA per step, weights from L2)
  k_lstm<<<BATCH / 16, 256, 0, stream>>>(x, b_ih, b_hh, wcomb, hs, lastf);

  // 3. fused proj + attention (online softmax) + out_proj + tanh*pi
  k_attn<<<BATCH / 16, 256, 0, stream>>>(s, lastf, projWb, proj_b, inpWb, inp_b,
                                         outWb, out_b, hs, merged, snorm);

  // 4. quantum circuit per batch row
  k_vqc<<<BATCH, 256, 0, stream>>>(snorm, qw, qfeat);

  // 5. MLP head
  k_pack<<<(BATCH * K1P + 255) / 256, 256, 0, stream>>>(merged, qfeat, mpack);
  k_mlp1<<<(32 * 32) / 8, 256, 0, stream>>>(mpack, w1p, b1, hid);
  k_mlp2<<<(BATCH * 8 + 255) / 256, 256, 0, stream>>>(hid, W2, b2, out);
}
